// CRF_60962765799643
// MI455X (gfx1250) — matrix operand order repaired and checked by reading, not compiled, not run on hardware
//
#include <hip/hip_runtime.h>
#include <stdint.h>

// ---------------------------------------------------------------------------
// CRF loss for MI455X (gfx1250).
//
// Forward algorithm recast as scaled exp-domain GEMM so it runs on the WMMA
// pipe:  alpha' = log( exp(alpha - rowmax) @ exp(trans) ) + rowmax + emit.
// E = exp(trans) is step-invariant and lives in LDS (f16-pair packed, 161KB
// of the 320KB WGP LDS).  16 workgroups x 17 waves; each WG owns 16 batch
// rows, each wave one 16-column N-tile; v_wmma_f32_16x16x32_f16, K=288 (9
// WMMAs per wave per step), 512 sequential steps with 3 barriers each.
// ---------------------------------------------------------------------------

#define NUM_TAGS 256
#define T2       258          // NUM_TAGS + 2 (start=256, end=257)
#define SEQ      512
#define BATCH    256
#define NEG      (-10000.0f)

#define KP     288            // K padded to multiple of 32 (258 -> 288)
#define KPW    144            // KP/2: packed f16-pair words along K
#define STR    148            // row stride in dwords: 16B aligned, bank-conflict free
#define NP     272            // N padded to 17 tiles of 16 (258 -> 272)
#define MB     16             // batch rows per workgroup
#define NWAVES 17             // one wave per 16-col N tile
#define THREADS (NWAVES * 32)

typedef __attribute__((ext_vector_type(16))) _Float16 v16h;
typedef __attribute__((ext_vector_type(8)))  float    v8f;

// LDS partition (dwords)
#define EPL_OFF   0                        // E packed:  [NP][STR]
#define PPL_OFF   (EPL_OFF + NP * STR)     // p packed:  [MB][STR]
#define ALPHA_OFF (PPL_OFF + MB * STR)     // alpha f32: [MB][KP]
#define RMAX_OFF  (ALPHA_OFF + MB * KP)    // rowmax:    [MB]
#define MLDS_OFF  (RMAX_OFF + MB)          // mask:      [MB]
#define LDS_WORDS (MLDS_OFF + MB)
#define LDS_BYTES (LDS_WORDS * 4)          // = 189,056 B  (< 320KB WGP LDS)

// ---------------------------------------------------------------------------
// Gold path score: 32 lanes per batch row, 16 steps each, wave32 reduction.
// ---------------------------------------------------------------------------
__global__ __launch_bounds__(256, 1)
void crf_gold_kernel(const float* __restrict__ em,
                     const unsigned char* __restrict__ masks,   // jax bool = 1 byte
                     const int* __restrict__ labels,
                     const float* __restrict__ trans,
                     float* __restrict__ gold)
{
    const int lane = threadIdx.x & 31;
    const int b    = blockIdx.x * 8 + (threadIdx.x >> 5);
    const int s0   = lane * 16;

    // incoming "cur" = most recent masked label before s0 (start tag if none)
    int cur = NUM_TAGS;
    for (int sb = s0 - 1; sb >= 0; --sb) {
        if (masks[b * SEQ + sb]) {
            int l = labels[b * SEQ + sb];
            cur = min(max(l, 0), NUM_TAGS - 1);
            break;
        }
    }

    float score = 0.0f;
    for (int s = s0; s < s0 + 16; ++s) {
        int mk = masks[b * SEQ + s];
        int l  = labels[b * SEQ + s];
        int cl = min(max(l, 0), NUM_TAGS - 1);
        if (mk) {
            score += em[((size_t)b * SEQ + s) * NUM_TAGS + cl] + trans[cur * T2 + cl];
            cur = cl;
        }
    }

    float tot = score;
    for (int off = 16; off >= 1; off >>= 1) tot += __shfl_xor(tot, off, 32);
    int curLast = __shfl(cur, 31, 32);               // lane31 holds final label
    if (lane == 0) gold[b] = tot + trans[curLast * T2 + (T2 - 1)];
}

// ---------------------------------------------------------------------------
// Forward (log-partition) via WMMA, then out[b] = gold[b] - fwd[b].
// ---------------------------------------------------------------------------
__global__ __launch_bounds__(THREADS, 1)
void crf_forward_kernel(const float* __restrict__ em,
                        const unsigned char* __restrict__ masks,
                        const float* __restrict__ trans,
                        const float* __restrict__ gold,
                        float* __restrict__ out)
{
    extern __shared__ __align__(16) unsigned int smem[];
    unsigned int* EPL = smem + EPL_OFF;          // E[k][n] packed {f16(E[2kp][n]), f16(E[2kp+1][n])}, laid [n][kp]
    unsigned int* PPL = smem + PPL_OFF;          // p packed same way,            laid [m][kp]
    float* ALPHA = (float*)(smem + ALPHA_OFF);   // [MB][KP]
    float* RMAX  = (float*)(smem + RMAX_OFF);    // [MB]
    float* MLDS  = (float*)(smem + MLDS_OFF);    // [MB]

    const int tid  = threadIdx.x;
    const int wave = tid >> 5;
    const int lane = tid & 31;
    const int b0   = blockIdx.x * MB;

    // ---- one-time: E = exp(trans), f16-pair packed into LDS (zero-padded) ----
    for (int idx = tid; idx < NP * KPW; idx += THREADS) {
        int n  = idx / KPW;           // destination column j (0..271)
        int kp = idx - n * KPW;       // packed K-pair index (0..143)
        int i0 = 2 * kp, i1 = i0 + 1;
        float e0 = 0.0f, e1 = 0.0f;
        if (n < T2) {
            if (i0 < T2) e0 = __expf(trans[i0 * T2 + n]);   // exp(-10000) == 0
            if (i1 < T2) e1 = __expf(trans[i1 * T2 + n]);
        }
        union { _Float16 h[2]; unsigned int u; } pk;
        pk.h[0] = (_Float16)e0; pk.h[1] = (_Float16)e1;
        EPL[n * STR + kp] = pk.u;
    }
    // ---- alpha0: NEG everywhere except start tag ----
    for (int idx = tid; idx < MB * KP; idx += THREADS) {
        int m = idx / KP, j = idx - m * KP;
        ALPHA[m * KP + j] = (j == NUM_TAGS) ? 0.0f : NEG;
    }
    __syncthreads();

    const int kh   = lane >> 4;                  // half-wave selects K sub-range
    const int mrow = lane & 15;                  // A fragment row
    const int ncol = wave * 16 + (lane & 15);    // B fragment column (tile = wave)

    for (int s = 0; s < SEQ; ++s) {
        // phase 1: rowmax (waves 0..15, one row each) + mask preload (wave 16)
        if (wave < MB) {
            const float* arow = ALPHA + wave * KP;
            float v = -1.0e30f;
            for (int j = lane; j < KP; j += 32) v = fmaxf(v, arow[j]);
            for (int off = 16; off >= 1; off >>= 1) v = fmaxf(v, __shfl_xor(v, off, 32));
            if (lane == 0) RMAX[wave] = v;
        } else if (lane < MB) {
            MLDS[lane] = masks[(size_t)(b0 + lane) * SEQ + s] ? 1.0f : 0.0f;
        }
        __syncthreads();

        // phase 2: p = exp(alpha - rowmax), f16-pair packed
        for (int idx = tid; idx < MB * KPW; idx += THREADS) {
            int m = idx / KPW, kp = idx - m * KPW;
            float rm = RMAX[m];
            float a0 = ALPHA[m * KP + 2 * kp];
            float a1 = ALPHA[m * KP + 2 * kp + 1];
            union { _Float16 h[2]; unsigned int u; } pk;
            pk.h[0] = (_Float16)__expf(a0 - rm);
            pk.h[1] = (_Float16)__expf(a1 - rm);
            PPL[m * STR + kp] = pk.u;
        }
        __syncthreads();

        // phase 3: acc[16x16] = p[16x288] @ E[288x(tile 16)] via 9 WMMAs
        v8f acc = (v8f){0.f, 0.f, 0.f, 0.f, 0.f, 0.f, 0.f, 0.f};
        const unsigned int* pA = &PPL[mrow * STR];
        const unsigned int* pB = &EPL[ncol * STR];
        #pragma unroll
        for (int kk = 0; kk < KPW; kk += 16) {   // 16 packed words == K step of 32
            union { v16h v; uint4 q[2]; } Af, Bf;
            // A (16x32 f16): VGPR v<4 -> K = kh*8 + 2v; v>=4 -> K = 16 + kh*8 + 2(v-4)
            Af.q[0] = *(const uint4*)(pA + kk + kh * 4);
            Af.q[1] = *(const uint4*)(pA + kk + kh * 4 + 8);
            // B (32x16 f16): VGPR v<4 -> K = kh*8 + 2v; v>=4 -> K = 16 + kh*8 + 2(v-4)
            Bf.q[0] = *(const uint4*)(pB + kk + kh * 4);
            Bf.q[1] = *(const uint4*)(pB + kk + kh * 4 + 8);
            acc = __builtin_amdgcn_wmma_f32_16x16x32_f16(
                false, Af.v, false, Bf.v, (short)0, acc, false, false);
        }

        // phase 4: alpha' = log(acc) + rowmax + emit, masked writeback
        {
            const int j = wave * 16 + (lane & 15);
            #pragma unroll
            for (int r = 0; r < 8; ++r) {
                const int m = r + 8 * kh;        // C/D layout: VGPR r, half kh
                if (j < T2) {
                    float emit = (j < NUM_TAGS)
                        ? em[((size_t)(b0 + m) * SEQ + s) * NUM_TAGS + j] : NEG;
                    float val = __logf(fmaxf(acc[r], 1e-35f)) + RMAX[m] + emit;
                    if (MLDS[m] != 0.0f) ALPHA[m * KP + j] = val;
                }
            }
        }
        __syncthreads();   // alpha stable before next rowmax
    }

    // ---- fwd = logsumexp_j(alpha[j] + trans[j,end]);  out = gold - fwd ----
    if (wave < MB) {
        const int b = b0 + wave;
        const float* arow = ALPHA + wave * KP;
        float mx = -1.0e30f;
        for (int j = lane; j < T2; j += 32)
            mx = fmaxf(mx, arow[j] + trans[j * T2 + (T2 - 1)]);
        for (int off = 16; off >= 1; off >>= 1) mx = fmaxf(mx, __shfl_xor(mx, off, 32));
        float sum = 0.0f;
        for (int j = lane; j < T2; j += 32)
            sum += __expf(arow[j] + trans[j * T2 + (T2 - 1)] - mx);
        for (int off = 16; off >= 1; off >>= 1) sum += __shfl_xor(sum, off, 32);
        if (lane == 0) out[b] = gold[b] - (__logf(sum) + mx);
    }
}

// ---------------------------------------------------------------------------
extern "C" void kernel_launch(void* const* d_in, const int* in_sizes, int n_in,
                              void* d_out, int out_size, void* d_ws, size_t ws_size,
                              hipStream_t stream)
{
    const float*         em     = (const float*)d_in[0];          // [256,512,256] f32
    const unsigned char* masks  = (const unsigned char*)d_in[1];  // [256,512] bool (1B)
    const int*           labels = (const int*)d_in[2];            // [256,512] i32
    const float*         trans  = (const float*)d_in[3];          // [258,258] f32
    float* out  = (float*)d_out;                                  // [256] f32
    float* gold = (float*)d_ws;                                   // scratch: 256 f32

    (void)in_sizes; (void)n_in; (void)out_size; (void)ws_size;

    crf_gold_kernel<<<BATCH / 8, 256, 0, stream>>>(em, masks, labels, trans, gold);

    // 189KB dynamic LDS: raise the limit (host-side, capture-safe, deterministic)
    (void)hipFuncSetAttribute((const void*)crf_forward_kernel,
                              hipFuncAttributeMaxDynamicSharedMemorySize, LDS_BYTES);
    crf_forward_kernel<<<BATCH / MB, THREADS, LDS_BYTES, stream>>>(
        em, masks, trans, gold, out);
}
